// Generator_30279519436852
// MI455X (gfx1250) — matrix operand order repaired and checked by reading, not compiled, not run on hardware
//
#include <hip/hip_runtime.h>
#include <math.h>

typedef __attribute__((ext_vector_type(16))) _Float16 v16h;
typedef __attribute__((ext_vector_type(8)))  float    v8f;

#define D   128
#define KNN 8

__device__ __forceinline__ float frelu(float x) { return x > 0.f ? x : 0.f; }

// ---------------------------------------------------------------------------
// (b,3,n) -> (b*n,3)
__global__ void k_transpose_in(const float* __restrict__ xyz, float* __restrict__ x,
                               int n, int total) {
  int gid = blockIdx.x * 256 + threadIdx.x;
  if (gid >= total) return;
  int j = gid % 3;
  int i = (gid / 3) % n;
  int b = gid / (3 * n);
  x[gid] = xyz[(size_t)b * 3 * n + (size_t)j * n + i];
}

// (b*n,3) -> (b,3,n)
__global__ void k_transpose_out(const float* __restrict__ xin, float* __restrict__ out,
                                int n, int total) {
  int gid = blockIdx.x * 256 + threadIdx.x;
  if (gid >= total) return;
  int m = gid % n;
  int j = (gid / n) % 3;
  int b = gid / (3 * n);
  out[gid] = xin[((size_t)b * n + m) * 3 + j];
}

// ---------------------------------------------------------------------------
// Brute-force kNN with LDS ref tiling and register top-8 insertion sort.
__global__ __launch_bounds__(256)
void k_knn(const float* __restrict__ q, const float* __restrict__ r,
           int nq, int nr, int* __restrict__ oidx) {
  const int b  = blockIdx.y;
  const int qi = blockIdx.x * 256 + threadIdx.x;
  const float* Q = q + (size_t)b * nq * 3;
  const float* R = r + (size_t)b * nr * 3;

  float qx = 0.f, qy = 0.f, qz = 0.f;
  if (qi < nq) { qx = Q[qi * 3]; qy = Q[qi * 3 + 1]; qz = Q[qi * 3 + 2]; }

  float bd[KNN]; int bi[KNN];
#pragma unroll
  for (int t = 0; t < KNN; ++t) { bd[t] = 3.0e38f; bi[t] = 0; }

  __shared__ float rs[256 * 3];
  for (int tile = 0; tile < nr; tile += 256) {
    int cnt = nr - tile; if (cnt > 256) cnt = 256;
    int li = threadIdx.x;
    if (li < cnt) {
      rs[li * 3 + 0] = R[(size_t)(tile + li) * 3 + 0];
      rs[li * 3 + 1] = R[(size_t)(tile + li) * 3 + 1];
      rs[li * 3 + 2] = R[(size_t)(tile + li) * 3 + 2];
    }
    __syncthreads();
    for (int jj = 0; jj < cnt; ++jj) {
      float dx = qx - rs[jj * 3], dy = qy - rs[jj * 3 + 1], dz = qz - rs[jj * 3 + 2];
      float d2 = dx * dx + dy * dy + dz * dz;
      if (d2 < bd[KNN - 1]) {
        int id = tile + jj;
#pragma unroll
        for (int t = KNN - 1; t >= 0; --t) {
          bool up = (t > 0) && (d2 < bd[t - 1]);
          if (d2 < bd[t]) {
            if (up) { bd[t] = bd[t - 1]; bi[t] = bi[t - 1]; }
            else    { bd[t] = d2;        bi[t] = id;        }
          }
        }
      }
    }
    __syncthreads();
  }
  if (qi < nq) {
#pragma unroll
    for (int t = 0; t < KNN; ++t)
      oidx[((size_t)b * nq + qi) * KNN + t] = bi[t];
  }
}

// ---------------------------------------------------------------------------
// Fused feature_net: per-point 8-neighbor rel coords -> 3 MLP layers -> max_k.
// Gather-dominated (16% of FLOPs): stays on VALU with LDS staging.
__global__ __launch_bounds__(128)
void k_feature_net(const float* __restrict__ x, const int* __restrict__ idx,
                   const float* __restrict__ w0, const float* __restrict__ w1,
                   const float* __restrict__ w2, float* __restrict__ o, int n) {
  const int pt   = blockIdx.x;
  const int c    = threadIdx.x;
  const int base = (pt / n) * n;
  __shared__ float g[KNN][3];
  __shared__ float h0[KNN][D];
  __shared__ float h1[KNN][D];
  if (c < KNN * 3) {
    int k = c / 3, j = c % 3;
    int nb = base + idx[(size_t)pt * KNN + k];
    g[k][j] = x[(size_t)nb * 3 + j] - x[(size_t)pt * 3 + j];
  }
  __syncthreads();
#pragma unroll
  for (int k = 0; k < KNN; ++k)
    h0[k][c] = frelu(g[k][0] * w0[c] + g[k][1] * w0[D + c] + g[k][2] * w0[2 * D + c]);
  __syncthreads();
  for (int k = 0; k < KNN; ++k) {
    float a = 0.f;
    for (int d = 0; d < D; ++d) a += h0[k][d] * w1[d * D + c];
    h1[k][c] = frelu(a);
  }
  __syncthreads();
  float m = -3.0e38f;
  for (int k = 0; k < KNN; ++k) {
    float a = 0.f;
    for (int d = 0; d < D; ++d) a += h1[k][d] * w2[d * D + c];
    m = fmaxf(m, frelu(a));
  }
  o[(size_t)pt * D + c] = m;
}

// ---------------------------------------------------------------------------
// p = relu(points); s = sum over 8 neighbors of relu(points)
__global__ __launch_bounds__(128)
void k_relu_gather(const float* __restrict__ pts, const int* __restrict__ idx,
                   float* __restrict__ p, float* __restrict__ s, int n) {
  const int pt   = blockIdx.x;
  const int c    = threadIdx.x;
  const int base = (pt / n) * n;
  float pv = frelu(pts[(size_t)pt * D + c]);
  p[(size_t)pt * D + c] = pv;
  float a = 0.f;
#pragma unroll
  for (int k = 0; k < KNN; ++k) {
    int nb = base + idx[(size_t)pt * KNN + k];
    a += frelu(pts[(size_t)nb * D + c]);
  }
  s[(size_t)pt * D + c] = a;
}

// ---------------------------------------------------------------------------
// Core WMMA GEMM: Out = (Ap @ Wc + As @ Wn) * scale + Sc
// Treated as one K=256 GEMM:  A' = [Ap | As] (rows x 256),  B' = [Wc ; Wn].
// B' is converted to f16 and staged in LDS pre-swizzled into wmma B-fragment
// layout (64 KB). 8 waves/block, each owning a 16x128 output stripe.
__global__ __launch_bounds__(256)
void k_gemm_dual(const float* __restrict__ Ap, const float* __restrict__ As,
                 const float* __restrict__ Wc, const float* __restrict__ Wn,
                 const float* __restrict__ Sc, float* __restrict__ Out,
                 float scale) {
  __shared__ _Float16 Bf[32768];  // 8 ktiles * 8 ntiles * 32 lanes * 16 halfs
  const int tid = threadIdx.x;

  // Cooperative fill of B fragments:
  // lane<16: N = j*16+lane,    K = t*32 + {0..7, 16..23}
  // lane>=16: N = j*16+lane-16, K = t*32 + {8..15, 24..31}
  for (int qq = tid; qq < 32768; qq += 256) {
    int e    = qq & 15;
    int lane = (qq >> 4) & 31;
    int tile = qq >> 9;            // 0..63
    int t = tile >> 3, j = tile & 7;
    int N = j * 16 + (lane & 15);
    int K = t * 32 + ((lane >> 4) * 8) + ((e < 8) ? e : 8 + e);
    float v = (K < 128) ? Wc[(size_t)K * D + N] : Wn[(size_t)(K - 128) * D + N];
    Bf[qq] = (_Float16)v;
  }
  __syncthreads();

  const int lane    = tid & 31;
  const int wave    = tid >> 5;
  const int rowbase = blockIdx.x * 128 + wave * 16;
  const int arow    = rowbase + (lane & 15);
  const int sel8    = (lane >> 4) * 8;   // K-half offset per documented A layout

  v8f zero = {};
  v8f acc[8];
#pragma unroll
  for (int j = 0; j < 8; ++j) acc[j] = zero;

#pragma unroll
  for (int t = 0; t < 8; ++t) {
    const float* src  = (t < 4) ? Ap : As;
    const int    kb   = (t & 3) * 32;
    const float* a    = src + (size_t)arow * D + kb + sel8;
    // A fragment, 16-bit A 16x32 layout:
    // lane<16 : K = kb + {0..7, 16..23};  lane>=16 : K = kb + {8..15, 24..31}
    float4 x0 = *reinterpret_cast<const float4*>(a);
    float4 x1 = *reinterpret_cast<const float4*>(a + 4);
    float4 y0 = *reinterpret_cast<const float4*>(a + 16);
    float4 y1 = *reinterpret_cast<const float4*>(a + 20);
    if (t < 7) {
      const float* nsrc = ((t + 1) < 4) ? Ap : As;
      __builtin_prefetch(nsrc + (size_t)arow * D + ((t + 1) & 3) * 32 + sel8);
    }
    v16h af;
    af[0]  = (_Float16)x0.x; af[1]  = (_Float16)x0.y; af[2]  = (_Float16)x0.z; af[3]  = (_Float16)x0.w;
    af[4]  = (_Float16)x1.x; af[5]  = (_Float16)x1.y; af[6]  = (_Float16)x1.z; af[7]  = (_Float16)x1.w;
    af[8]  = (_Float16)y0.x; af[9]  = (_Float16)y0.y; af[10] = (_Float16)y0.z; af[11] = (_Float16)y0.w;
    af[12] = (_Float16)y1.x; af[13] = (_Float16)y1.y; af[14] = (_Float16)y1.z; af[15] = (_Float16)y1.w;
#pragma unroll
    for (int j = 0; j < 8; ++j) {
      const v16h bf = *reinterpret_cast<const v16h*>(&Bf[(((t * 8 + j) * 32) + lane) * 16]);
      acc[j] = __builtin_amdgcn_wmma_f32_16x16x32_f16(
          /*neg_a=*/false, af, /*neg_b=*/false, bf,
          /*c_mod=*/(short)0, acc[j], /*reuse_a=*/false, /*reuse_b=*/false);
    }
  }

  // Epilogue per documented f32 C layout: element r -> M = r (+8 if lane>=16), N = lane&15
  const int rowm  = rowbase + ((lane < 16) ? 0 : 8);
  const int ncol0 = lane & 15;
#pragma unroll
  for (int j = 0; j < 8; ++j) {
#pragma unroll
    for (int r = 0; r < 8; ++r) {
      size_t row = (size_t)(rowm + r);
      int    col = j * 16 + ncol0;
      Out[row * D + col] = acc[j][r] * scale + Sc[row * D + col];
    }
  }
}

// ---------------------------------------------------------------------------
// off[pt][e] = (p[pt]·uc[:,e] + s[pt]·un[:,e]) / 9,  e < 6
__global__ void k_offset6(const float* __restrict__ p, const float* __restrict__ s,
                          const float* __restrict__ uc, const float* __restrict__ un,
                          float* __restrict__ off, int rows) {
  int gid = blockIdx.x * 256 + threadIdx.x;
  if (gid >= rows * 6) return;
  int pt = gid / 6, e = gid % 6;
  float a = 0.f;
  for (int d = 0; d < D; ++d)
    a += p[(size_t)pt * D + d] * uc[d * 6 + e] + s[(size_t)pt * D + d] * un[d * 6 + e];
  off[gid] = a * (1.f / 9.f);
}

// new_xyz[b][r*n + i][j] = x[b][i][j] + off[b][i][j*2 + r]
__global__ void k_unpool(const float* __restrict__ x, const float* __restrict__ off,
                         float* __restrict__ nx, int n, int total6) {
  int gid = blockIdx.x * 256 + threadIdx.x;
  if (gid >= total6) return;
  int pt = gid / 6, e = gid % 6;
  int j = e / 2, r = e % 2;
  int bb = pt / n, local = pt % n;
  nx[((size_t)bb * 2 * n + (size_t)r * n + local) * 3 + j] =
      x[(size_t)pt * 3 + j] + off[(size_t)pt * 6 + e];
}

// points2[m][c] = mean_k points[base + idx[m][k]][c]
__global__ __launch_bounds__(128)
void k_gather_mean(const float* __restrict__ pts, const int* __restrict__ idx,
                   float* __restrict__ o, int nq, int nr) {
  int m = blockIdx.x;
  int c = threadIdx.x;
  int base = (m / nq) * nr;
  float a = 0.f;
#pragma unroll
  for (int k = 0; k < KNN; ++k)
    a += pts[((size_t)base + idx[(size_t)m * KNN + k]) * D + c];
  o[(size_t)m * D + c] = a * (1.f / KNN);
}

// ---------------------------------------------------------------------------
extern "C" void kernel_launch(void* const* d_in, const int* in_sizes, int n_in,
                              void* d_out, int out_size, void* d_ws, size_t ws_size,
                              hipStream_t stream) {
  (void)in_sizes; (void)n_in; (void)out_size; (void)ws_size;
  const float* xyz  = (const float*)d_in[0];
  const float* w0   = (const float*)d_in[1];
  const float* w1   = (const float*)d_in[2];
  const float* w2   = (const float*)d_in[3];
  const float* u1wc = (const float*)d_in[4];
  const float* u1wn = (const float*)d_in[5];
  const float* u1uc = (const float*)d_in[6];
  const float* u1un = (const float*)d_in[7];
  const float* u2wc = (const float*)d_in[8];
  const float* u2wn = (const float*)d_in[9];
  const float* u2uc = (const float*)d_in[10];
  const float* u2un = (const float*)d_in[11];
  float* out = (float*)d_out;

  const int B = 4, N1 = 4096, N2 = 8192, N4 = 16384;

  char* wp = (char*)d_ws;
  auto alloc = [&](size_t bytes) -> char* {
    char* p = wp;
    wp += (bytes + 255) & ~(size_t)255;
    return p;
  };
  float* x    = (float*)alloc((size_t)B * N1 * 3 * 4);
  float* nx1  = (float*)alloc((size_t)B * N2 * 3 * 4);
  float* nx2  = (float*)alloc((size_t)B * N4 * 3 * 4);
  int*   idx1 = (int*)  alloc((size_t)B * N1 * KNN * 4);
  int*   idx2 = (int*)  alloc((size_t)B * N2 * KNN * 4);
  int*   idx3 = (int*)  alloc((size_t)B * N2 * KNN * 4);
  float* ptsA = (float*)alloc((size_t)B * N2 * D * 4);
  float* ptsB = (float*)alloc((size_t)B * N2 * D * 4);
  float* pbuf = (float*)alloc((size_t)B * N2 * D * 4);
  float* sbuf = (float*)alloc((size_t)B * N2 * D * 4);
  float* off  = (float*)alloc((size_t)B * N2 * 6 * 4);

  // 1) transpose xyz (b,3,n) -> x (b*n,3)
  { int total = B * N1 * 3;
    k_transpose_in<<<(total + 255) / 256, 256, 0, stream>>>(xyz, x, N1, total); }

  // 2) self-kNN on x (shared by feature_net and stage-1 blocks)
  { dim3 g((N1 + 255) / 256, B);
    k_knn<<<g, 256, 0, stream>>>(x, x, N1, N1, idx1); }

  // 3) feature_net -> ptsA (B*N1 x 128)
  k_feature_net<<<B * N1, D, 0, stream>>>(x, idx1, w0, w1, w2, ptsA, N1);

  // 4) stage-1: 12 residual graph-conv blocks (WMMA GEMM, K=256)
  float* cur = ptsA; float* nxt = ptsB;
  for (int i = 0; i < 12; ++i) {
    k_relu_gather<<<B * N1, D, 0, stream>>>(cur, idx1, pbuf, sbuf, N1);
    k_gemm_dual<<<(B * N1) / 128, 256, 0, stream>>>(
        pbuf, sbuf, u1wc + (size_t)i * D * D, u1wn + (size_t)i * D * D,
        cur, nxt, 1.f / 9.f);
    float* t = cur; cur = nxt; nxt = t;
    if (i == 11) {
      int total6 = B * N1 * 6;
      k_offset6<<<(total6 + 255) / 256, 256, 0, stream>>>(pbuf, sbuf, u1uc, u1un, off, B * N1);
      k_unpool<<<(total6 + 255) / 256, 256, 0, stream>>>(x, off, nx1, N1, total6);
    }
  }

  // 5) kNN of new points among original points; gather + mean
  { dim3 g((N2 + 255) / 256, B);
    k_knn<<<g, 256, 0, stream>>>(nx1, x, N2, N1, idx2); }
  k_gather_mean<<<B * N2, D, 0, stream>>>(cur, idx2, nxt, N2, N1);
  { float* t = cur; cur = nxt; nxt = t; }   // cur now (B*N2, 128)

  // 6) self-kNN on nx1
  { dim3 g((N2 + 255) / 256, B);
    k_knn<<<g, 256, 0, stream>>>(nx1, nx1, N2, N2, idx3); }

  // 7) stage-2: 12 blocks at 2n
  for (int i = 0; i < 12; ++i) {
    k_relu_gather<<<B * N2, D, 0, stream>>>(cur, idx3, pbuf, sbuf, N2);
    k_gemm_dual<<<(B * N2) / 128, 256, 0, stream>>>(
        pbuf, sbuf, u2wc + (size_t)i * D * D, u2wn + (size_t)i * D * D,
        cur, nxt, 1.f / 9.f);
    float* t = cur; cur = nxt; nxt = t;
    if (i == 11) {
      int total6 = B * N2 * 6;
      k_offset6<<<(total6 + 255) / 256, 256, 0, stream>>>(pbuf, sbuf, u2uc, u2un, off, B * N2);
      k_unpool<<<(total6 + 255) / 256, 256, 0, stream>>>(nx1, off, nx2, N2, total6);
    }
  }

  // 8) transpose (b*4n,3) -> (b,3,4n)
  { int total = B * N4 * 3;
    k_transpose_out<<<(total + 255) / 256, 256, 0, stream>>>(nx2, out, N4, total); }
}
